// GAT_33663953666346
// MI455X (gfx1250) — matrix operand order repaired and checked by reading, not compiled, not run on hardware
//
#include <hip/hip_runtime.h>
#include <cmath>

// ---------- types ----------
typedef __attribute__((ext_vector_type(16))) __bf16        v16bf;
typedef __attribute__((ext_vector_type(8)))  float         v8f;
typedef __attribute__((ext_vector_type(4)))  unsigned int  v4u;

#define NNODES 50000
#define NEDGES 800000
#define INDIM  256
#define HIDC   32
#define HEADS  8
#define OUTDIM 64
#define KGEMM  256                       // K is 256 for all four GEMMs
#define BSTRIDE (KGEMM + 8)              // padded LDS column stride (elements)
#define GEMM_LDS_BYTES (16 * BSTRIDE * 2)

// ---------- small utility kernels ----------
__global__ void fill_f32(float* __restrict__ p, long long n, float v) {
  long long i = (long long)blockIdx.x * blockDim.x + threadIdx.x;
  if (i < n) p[i] = v;
}

__global__ void cvt_bf16(const float* __restrict__ in, __bf16* __restrict__ out, long long n) {
  long long i = (long long)blockIdx.x * blockDim.x + threadIdx.x;
  if (i < n) out[i] = (__bf16)in[i];
}

// W[k][n] (row-major K x Nc, f32) -> Wt[n][k] (col-major, bf16, K contiguous)
__global__ void transpose_cvt(const float* __restrict__ W, __bf16* __restrict__ Wt,
                              int K, int Nc) {
  int i = blockIdx.x * blockDim.x + threadIdx.x;
  if (i >= K * Nc) return;
  int k = i / Nc;
  int n = i - k * Nc;
  Wt[(size_t)n * K + k] = (__bf16)W[i];
}

// ---------- WMMA GEMM: C[M x Nc] = A_bf16[M x 256] @ B + bias, B given as Bt[n][k] ----
// Block = 8 waves, one 16x16 C tile per wave (8 M-tiles x 1 N-tile per block).
// The block's shared 16-col B tile (16x256 bf16 = 8KB) is staged to LDS once with
// global_load_async_to_lds_b128 (ASYNCcnt), padded to kill ds bank conflicts, and
// re-read by all 8 waves (8x cut in B global traffic). A streams from global/L2.
__global__ void __launch_bounds__(256)
gemm_bf16_wmma(const __bf16* __restrict__ A, const __bf16* __restrict__ Bt,
               const float* __restrict__ bias, float* __restrict__ C,
               int M, int Nc) {
  extern __shared__ __bf16 Bs[];                   // dynamic LDS: base offset 0
  const int lane  = threadIdx.x & 31;
  const int wave  = threadIdx.x >> 5;
  const int mtile = blockIdx.y * 8 + wave;
  const int col0  = blockIdx.x << 4;

  // ---- async stage B tile: 512 chunks of 16B, 2 per thread ----
#pragma unroll
  for (int it = 0; it < 2; ++it) {
    const int c   = threadIdx.x + (it << 8);       // 0..511
    const int col = c >> 5;                        // 16 columns
    const int k8  = c & 31;                        // 32 x 8-element chunks per column
    const unsigned lds = (unsigned)(col * (BSTRIDE * 2) + k8 * 16);
    const unsigned long long ga =
        (unsigned long long)(Bt + (size_t)(col0 + col) * KGEMM + k8 * 8);
    asm volatile("global_load_async_to_lds_b128 %0, %1, off"
                 :: "v"(lds), "v"(ga) : "memory");
  }
  asm volatile("s_wait_asynccnt 0x0" ::: "memory");
  __syncthreads();

  if (mtile < (M >> 4)) {                          // wave-uniform: EXEC all-ones for WMMA
    const int row0 = mtile << 4;
    const int half = lane >> 4;
    const int lr   = lane & 15;

    const __bf16* __restrict__ arow = A + (size_t)(row0 + lr) * KGEMM;
    const __bf16* bcol = Bs + lr * BSTRIDE;        // LDS, conflict-free (stride 132 dw)

    v8f acc = {};
#pragma unroll
    for (int kb = 0; kb < KGEMM; kb += 32) {
      __builtin_prefetch(arow + kb + 128, 0, 0);   // global_prefetch_b8
      union { v16bf v; v4u q[2]; } a, b;
      a.q[0] = *(const v4u*)(arow + kb + half * 8);
      a.q[1] = *(const v4u*)(arow + kb + 16 + half * 8);
      b.q[0] = *(const v4u*)(bcol + kb + half * 8);        // ds_load_b128
      b.q[1] = *(const v4u*)(bcol + kb + 16 + half * 8);
      acc = __builtin_amdgcn_wmma_f32_16x16x32_bf16(
          /*neg_a=*/false, a.v, /*neg_b=*/false, b.v,
          /*c_mod=*/(short)0, acc, /*reuse_a=*/false, /*reuse_b=*/false);
    }

    const float bs = bias[col0 + lr];
    // D layout: VGPR r -> (M = r + 8*half, N = lr)
#pragma unroll
    for (int r = 0; r < 8; ++r) {
      C[(size_t)(row0 + half * 8 + r) * Nc + col0 + lr] = acc[r] + bs;
    }
  }
}

// ---------- edge kernels ----------
__device__ __forceinline__ void atomicMaxF(float* addr, float v) {
  // IEEE-ordered max via signed/unsigned int atomics (init must be -inf)
  if (v >= 0.0f) atomicMax((int*)addr, __float_as_int(v));
  else           atomicMin((unsigned int*)addr, __float_as_uint(v));
}

// Wave per (edge, head), lane per channel (C == 32 == wave32).
// score[e*H+h] = sum_c att[h][c] * leaky_relu(xl[src][h*32+c] + xr[dst][h*32+c])
__global__ void edge_score(const int* __restrict__ src, const int* __restrict__ dst,
                           const float* __restrict__ xl, const float* __restrict__ xr,
                           const float* __restrict__ att, float* __restrict__ score,
                           float* __restrict__ smax, int E, int hsh) {
  const int H = 1 << hsh;
  const int widx = blockIdx.x * (blockDim.x >> 5) + (threadIdx.x >> 5);
  const int lane = threadIdx.x & 31;
  if (widx >= (E << hsh)) return;
  const int e = widx >> hsh;
  const int h = widx & (H - 1);
  const int s = src[e], d = dst[e];
  const int fo = (h << 5) + lane;
  float v = xl[(size_t)s * (H << 5) + fo] + xr[(size_t)d * (H << 5) + fo];
  v = v > 0.0f ? v : 0.2f * v;                       // leaky_relu, NEG_SLOPE=0.2
  float acc = att[fo] * v;
  acc += __shfl_xor(acc, 16, 32);
  acc += __shfl_xor(acc, 8, 32);
  acc += __shfl_xor(acc, 4, 32);
  acc += __shfl_xor(acc, 2, 32);
  acc += __shfl_xor(acc, 1, 32);
  if (lane == 0) {
    score[widx] = acc;
    atomicMaxF(&smax[(size_t)d * H + h], acc);
  }
}

// score <- exp(score - smax[dst]); denom[dst] += score (atomic)
__global__ void edge_exp(float* __restrict__ score, const float* __restrict__ smax,
                         float* __restrict__ denom, const int* __restrict__ dst,
                         int E, int hsh) {
  const int H = 1 << hsh;
  int idx = blockIdx.x * blockDim.x + threadIdx.x;
  if (idx >= (E << hsh)) return;
  const int e = idx >> hsh;
  const int h = idx & (H - 1);
  const int d = dst[e];
  float ex = __expf(score[idx] - smax[(size_t)d * H + h]);
  score[idx] = ex;
  atomicAdd(&denom[(size_t)d * H + h], ex);
}

// out[dst][f..f+3] += (score[e][h]/denom[dst][h]) * xl[src][f..f+3]  (float4 per thread)
__global__ void edge_agg(const float* __restrict__ score, const float* __restrict__ denom,
                         const float* __restrict__ xl,
                         const int* __restrict__ src, const int* __restrict__ dst,
                         float* __restrict__ out, int E, int hsh, int csh) {
  const int qsh = hsh + csh - 2;                  // log2(float4 groups per edge)
  const int Q   = 1 << qsh;
  const int H   = 1 << hsh;
  long long idx = (long long)blockIdx.x * blockDim.x + threadIdx.x;
  if (idx >= ((long long)E << qsh)) return;
  const int e = (int)(idx >> qsh);
  const int f = ((int)idx & (Q - 1)) << 2;
  const int h = f >> csh;
  const int s = src[e], d = dst[e];
  const float alpha = score[(size_t)e * H + h] / (denom[(size_t)d * H + h] + 1e-16f);
  const float4 xv = *(const float4*)(xl + ((size_t)s << (hsh + csh)) + f);
  float* po = out + ((size_t)d << (hsh + csh)) + f;
  atomicAdd(po + 0, alpha * xv.x);
  atomicAdd(po + 1, alpha * xv.y);
  atomicAdd(po + 2, alpha * xv.z);
  atomicAdd(po + 3, alpha * xv.w);
}

// h <- elu(h + bias[f]); optional bf16 copy for the next GEMM's A operand
__global__ void elu_bias(float* __restrict__ h, const float* __restrict__ bias,
                         __bf16* __restrict__ hbf, long long n, int Fmask) {
  long long i = (long long)blockIdx.x * blockDim.x + threadIdx.x;
  if (i >= n) return;
  float v = h[i] + bias[(int)i & Fmask];
  v = v > 0.0f ? v : (__expf(v) - 1.0f);          // elu
  h[i] = v;
  if (hbf) hbf[i] = (__bf16)v;
}

// out[n][o] = b[o] + sum_k h[n][k] * W[k][o]   (K=32, O=64, f32)
__global__ void final_linear(const float* __restrict__ h, const float* __restrict__ W,
                             const float* __restrict__ b, float* __restrict__ out, int n) {
  int idx = blockIdx.x * blockDim.x + threadIdx.x;
  if (idx >= n * OUTDIM) return;
  const int i = idx >> 6;
  const int o = idx & 63;
  const float* __restrict__ hr = h + (size_t)i * HIDC;
  float acc = b[o];
#pragma unroll 8
  for (int k = 0; k < HIDC; ++k) acc = fmaf(hr[k], W[k * OUTDIM + o], acc);
  out[idx] = acc;
}

// ---------- launcher ----------
extern "C" void kernel_launch(void* const* d_in, const int* in_sizes, int n_in,
                              void* d_out, int out_size, void* d_ws, size_t ws_size,
                              hipStream_t stream) {
  const float* x     = (const float*)d_in[0];
  const int*   ei    = (const int*)  d_in[1];
  const float* W1l   = (const float*)d_in[2];
  const float* b1l   = (const float*)d_in[3];
  const float* W1r   = (const float*)d_in[4];
  const float* b1r   = (const float*)d_in[5];
  const float* att1  = (const float*)d_in[6];
  const float* bias1 = (const float*)d_in[7];
  const float* W2l   = (const float*)d_in[8];
  const float* b2l   = (const float*)d_in[9];
  const float* W2r   = (const float*)d_in[10];
  const float* b2r   = (const float*)d_in[11];
  const float* att2  = (const float*)d_in[12];
  const float* bias2 = (const float*)d_in[13];
  const float* Wlin  = (const float*)d_in[14];
  const float* blin  = (const float*)d_in[15];
  float* out = (float*)d_out;

  const int* srcp = ei;
  const int* dstp = ei + NEDGES;

  // ---- workspace carve (256B aligned) ----
  char* p = (char*)d_ws;
  auto carve = [&](size_t bytes) -> void* {
    void* r = (void*)p;
    p += (bytes + 255) & ~(size_t)255;
    return r;
  };
  __bf16* Xbf  = (__bf16*)carve((size_t)NNODES * INDIM * 2);   // reused as h1 bf16
  __bf16* Wt1l = (__bf16*)carve((size_t)INDIM * (HEADS*HIDC) * 2);
  __bf16* Wt1r = (__bf16*)carve((size_t)INDIM * (HEADS*HIDC) * 2);
  __bf16* Wt2l = (__bf16*)carve((size_t)(HEADS*HIDC) * HIDC * 2);
  __bf16* Wt2r = (__bf16*)carve((size_t)(HEADS*HIDC) * HIDC * 2);
  float*  xl1  = (float*)carve((size_t)NNODES * 256 * 4);      // reused as xl2
  float*  xr1  = (float*)carve((size_t)NNODES * 256 * 4);      // reused as xr2 + h2
  float*  scr  = (float*)carve((size_t)NEDGES * HEADS * 4);    // scores/alphas (layer2 uses E)
  float*  smax = (float*)carve((size_t)NNODES * HEADS * 4);
  float*  den  = (float*)carve((size_t)NNODES * HEADS * 4);
  float*  h1   = (float*)carve((size_t)NNODES * 256 * 4);
  float*  xl2  = xl1;
  float*  xr2  = xr1;
  float*  h2   = xr1 + (size_t)NNODES * HIDC;                  // disjoint from xr2 region

  const int T = 256;
  auto blks = [](long long n, int t) { return (unsigned)((n + t - 1) / t); };

  // ---- prep: bf16 conversion + weight transpose ----
  cvt_bf16<<<blks((long long)NNODES*INDIM, T), T, 0, stream>>>(x, Xbf, (long long)NNODES*INDIM);
  transpose_cvt<<<blks(INDIM*256, T), T, 0, stream>>>(W1l, Wt1l, INDIM, 256);
  transpose_cvt<<<blks(INDIM*256, T), T, 0, stream>>>(W1r, Wt1r, INDIM, 256);
  transpose_cvt<<<blks(256*HIDC, T), T, 0, stream>>>(W2l, Wt2l, 256, HIDC);
  transpose_cvt<<<blks(256*HIDC, T), T, 0, stream>>>(W2r, Wt2r, 256, HIDC);

  // ---- layer 1: GATv2(256 -> 32, heads=8) ----
  fill_f32<<<blks((long long)NNODES*HEADS, T), T, 0, stream>>>(smax, (long long)NNODES*HEADS, -INFINITY);
  fill_f32<<<blks((long long)NNODES*HEADS, T), T, 0, stream>>>(den,  (long long)NNODES*HEADS, 0.0f);
  fill_f32<<<blks((long long)NNODES*256,  T), T, 0, stream>>>(h1,   (long long)NNODES*256,  0.0f);

  {
    dim3 g(256 / 16, (unsigned)(((NNODES >> 4) + 7) / 8), 1);
    gemm_bf16_wmma<<<g, 256, GEMM_LDS_BYTES, stream>>>(Xbf, Wt1l, b1l, xl1, NNODES, 256);
    gemm_bf16_wmma<<<g, 256, GEMM_LDS_BYTES, stream>>>(Xbf, Wt1r, b1r, xr1, NNODES, 256);
  }
  // wave per (edge,head): 8 waves per 256-thread block
  edge_score<<<blks((long long)NEDGES*HEADS, 8), T, 0, stream>>>(srcp, dstp, xl1, xr1, att1, scr, smax, NEDGES, 3);
  edge_exp  <<<blks((long long)NEDGES*HEADS, T), T, 0, stream>>>(scr, smax, den, dstp, NEDGES, 3);
  edge_agg  <<<blks((long long)NEDGES*64,    T), T, 0, stream>>>(scr, den, xl1, srcp, dstp, h1, NEDGES, 3, 5);
  // elu + bias, and write bf16 copy into Xbf for conv2's A operand
  elu_bias<<<blks((long long)NNODES*256, T), T, 0, stream>>>(h1, bias1, Xbf, (long long)NNODES*256, 255);

  // ---- layer 2: GATv2(256 -> 32, heads=1) ----
  fill_f32<<<blks((long long)NNODES, T), T, 0, stream>>>(smax, (long long)NNODES, -INFINITY);
  fill_f32<<<blks((long long)NNODES, T), T, 0, stream>>>(den,  (long long)NNODES, 0.0f);
  fill_f32<<<blks((long long)NNODES*HIDC, T), T, 0, stream>>>(h2, (long long)NNODES*HIDC, 0.0f);
  {
    dim3 g(HIDC / 16, (unsigned)(((NNODES >> 4) + 7) / 8), 1);
    gemm_bf16_wmma<<<g, 256, GEMM_LDS_BYTES, stream>>>(Xbf, Wt2l, b2l, xl2, NNODES, HIDC);
    gemm_bf16_wmma<<<g, 256, GEMM_LDS_BYTES, stream>>>(Xbf, Wt2r, b2r, xr2, NNODES, HIDC);
  }
  edge_score<<<blks((long long)NEDGES, 8), T, 0, stream>>>(srcp, dstp, xl2, xr2, att2, scr, smax, NEDGES, 0);
  edge_exp  <<<blks((long long)NEDGES, T), T, 0, stream>>>(scr, smax, den, dstp, NEDGES, 0);
  edge_agg  <<<blks((long long)NEDGES*8, T), T, 0, stream>>>(scr, den, xl2, srcp, dstp, h2, NEDGES, 0, 5);
  elu_bias<<<blks((long long)NNODES*HIDC, T), T, 0, stream>>>(h2, bias2, (__bf16*)nullptr, (long long)NNODES*HIDC, 31);

  // ---- final linear 32 -> 64 ----
  final_linear<<<blks((long long)NNODES*OUTDIM, T), T, 0, stream>>>(h2, Wlin, blin, out, NNODES);
}
